// AngularAttention_42262478193303
// MI455X (gfx1250) — matrix operand order repaired and checked by reading, not compiled, not run on hardware
//
#include <hip/hip_runtime.h>
#include <hip/hip_fp16.h>
#include <math.h>

typedef __attribute__((ext_vector_type(16))) _Float16 v16h;
typedef __attribute__((ext_vector_type(8)))  float    v8f;
typedef int v4i __attribute__((vector_size(16)));

#define D_IN      1024
#define D_OUT     1024
#define NUM_HEADS 16
#define HEAD_DIM  64
#define B_SZ      2
#define T_SZ      2048
#define N_ROWS    (B_SZ * T_SZ)
#define EPS       1e-6f
#define INV_PI    0.3183098861837907f

// ---------------------------------------------------------------------------
// CDNA5 async global->LDS staging (ASYNCcnt path), guarded so compile never
// breaks if the builtins are absent on this toolchain.
// Probe-learned signature: (AS(1) v4i* gsrc, AS(3) v4i* ldst, i32 off, i32 cpol)
// ---------------------------------------------------------------------------
#if defined(__gfx1250__) && __has_builtin(__builtin_amdgcn_global_load_async_to_lds_b128) && __has_builtin(__builtin_amdgcn_s_wait_asynccnt)
#define HAVE_ASYNC_LDS 1
#endif

__device__ __forceinline__ void copy_b128_to_lds(const _Float16* gsrc, _Float16* ldst) {
#ifdef HAVE_ASYNC_LDS
    // AS(3) pointers are the low-32-bit LDS offset (ISA aperture rule); AS(1)
    // global pointers keep the same 64-bit value. Route through uintptr_t.
    __builtin_amdgcn_global_load_async_to_lds_b128(
        (__attribute__((address_space(1))) v4i*)(uintptr_t)gsrc,
        (__attribute__((address_space(3))) v4i*)(uintptr_t)ldst, 0, 0);
#else
    *(uint4*)ldst = *(const uint4*)gsrc;
#endif
}
__device__ __forceinline__ void wait_async0() {
#ifdef HAVE_ASYNC_LDS
    __builtin_amdgcn_s_wait_asynccnt(0);
#endif
}

// ---------------------------------------------------------------------------
// WMMA fragment loaders (wave32; A/B use this code's contiguous k order, matched)
// A 16x32 (MxK): lane<16 -> M=lane, k-half 0; lane>=16 -> M=lane-16, k-half 1
//   element e -> K = e + hi*16
// B 32x16 (KxN): lane gives N (mod 16); K = e + hi*16
// C 16x16 f32:   lane gives N (mod 16); VGPR r -> M = r + hi*8
// ---------------------------------------------------------------------------
__device__ __forceinline__ v16h load_frag_A(const _Float16* __restrict__ base, int stride) {
    int lane = threadIdx.x & 31;
    int m  = lane & 15;
    int hi = (lane >> 4) & 1;
    v16h a;
#pragma unroll
    for (int e = 0; e < 16; ++e) {
        int k = e + (hi ? 16 : 0);
        a[e] = base[m * stride + k];
    }
    return a;
}

// B stored n-major in LDS: element (k,n) at base[n*stride + k]
__device__ __forceinline__ v16h load_frag_B_nmaj(const _Float16* __restrict__ base, int stride) {
    int lane = threadIdx.x & 31;
    int n  = lane & 15;
    int kh = (lane >> 4) ? 16 : 0;
    v16h b;
#pragma unroll
    for (int e = 0; e < 16; ++e) b[e] = base[n * stride + (e + kh)];
    return b;
}

// B stored k-major in LDS: element (k,n) at base[k*stride + n]
__device__ __forceinline__ v16h load_frag_B_kmaj(const _Float16* __restrict__ base, int stride) {
    int lane = threadIdx.x & 31;
    int n  = lane & 15;
    int kh = (lane >> 4) ? 16 : 0;
    v16h b;
#pragma unroll
    for (int e = 0; e < 16; ++e) b[e] = base[(e + kh) * stride + n];
    return b;
}

// ---------------------------------------------------------------------------
// f32 -> f16 conversion
// ---------------------------------------------------------------------------
__global__ void cvt_f32_f16(const float* __restrict__ src, _Float16* __restrict__ dst, int n) {
    int i = blockIdx.x * blockDim.x + threadIdx.x;
    int stride = gridDim.x * blockDim.x;
    for (; i < n; i += stride) dst[i] = (_Float16)src[i];
}

// ---------------------------------------------------------------------------
// GEMM: C[M,Nout] = A[M,K] * Bw[Nout,K]^T (+bias), f16 in / f32 out
// 128x128 block tile, 8 waves in 4x2, each wave 32x64 (2x4 WMMA tiles).
// Double-buffered LDS ping-pong: global loads for panel s+1 are issued
// before the 8 WMMAs of panel s; one barrier per K-step.
// ---------------------------------------------------------------------------
__launch_bounds__(256)
__global__ void gemm_f16_wmma(const _Float16* __restrict__ A, const _Float16* __restrict__ Bw,
                              float* __restrict__ C, const float* __restrict__ bias,
                              int M, int Nout, int K) {
    __shared__ _Float16 sA[2][128 * 32];
    __shared__ _Float16 sB[2][128 * 32];
    int tid  = threadIdx.x;
    int lane = tid & 31, wave = tid >> 5;
    int wm = wave & 3, wn = wave >> 2;       // 4x2 wave grid
    int hi = lane >> 4, nn = lane & 15;
    int m0 = blockIdx.y * 128;
    int n0 = blockIdx.x * 128;

    int lrow = tid >> 1;            // 0..127
    int lcol = (tid & 1) * 16;      // 0 or 16 (halfs); each thread: 2 uint4

    const _Float16* Ap = A  + (size_t)(m0 + lrow) * K + lcol;
    const _Float16* Bp = Bw + (size_t)(n0 + lrow) * K + lcol;

    v8f acc[2][4];
#pragma unroll
    for (int i = 0; i < 2; ++i)
#pragma unroll
        for (int j = 0; j < 4; ++j) acc[i][j] = (v8f){};

    uint4 ra0 = *(const uint4*)(Ap);
    uint4 ra1 = *(const uint4*)(Ap + 8);
    uint4 rb0 = *(const uint4*)(Bp);
    uint4 rb1 = *(const uint4*)(Bp + 8);
    *(uint4*)(sA[0] + lrow * 32 + lcol)     = ra0;
    *(uint4*)(sA[0] + lrow * 32 + lcol + 8) = ra1;
    *(uint4*)(sB[0] + lrow * 32 + lcol)     = rb0;
    *(uint4*)(sB[0] + lrow * 32 + lcol + 8) = rb1;
    __syncthreads();

    int nsteps = K >> 5;
    for (int s = 0; s < nsteps; ++s) {
        int cur = s & 1;
        if (s + 1 < nsteps) {
            const _Float16* An = Ap + (s + 1) * 32;
            const _Float16* Bn = Bp + (s + 1) * 32;
            ra0 = *(const uint4*)(An);
            ra1 = *(const uint4*)(An + 8);
            rb0 = *(const uint4*)(Bn);
            rb1 = *(const uint4*)(Bn + 8);
            if (s + 2 < nsteps) {                 // L2 prefetch of panel s+2
                __builtin_prefetch(An + 32, 0, 1);
                __builtin_prefetch(Bn + 32, 0, 1);
            }
        }
        v16h af[2], bf[4];
#pragma unroll
        for (int i = 0; i < 2; ++i)
            af[i] = load_frag_A(sA[cur] + (wm * 32 + i * 16) * 32, 32);
#pragma unroll
        for (int j = 0; j < 4; ++j)
            bf[j] = load_frag_B_nmaj(sB[cur] + (wn * 64 + j * 16) * 32, 32);
#pragma unroll
        for (int i = 0; i < 2; ++i)
#pragma unroll
            for (int j = 0; j < 4; ++j)
                acc[i][j] = __builtin_amdgcn_wmma_f32_16x16x32_f16(
                    false, af[i], false, bf[j], (short)0, acc[i][j], false, false);
        if (s + 1 < nsteps) {
            int nxt = cur ^ 1;
            *(uint4*)(sA[nxt] + lrow * 32 + lcol)     = ra0;
            *(uint4*)(sA[nxt] + lrow * 32 + lcol + 8) = ra1;
            *(uint4*)(sB[nxt] + lrow * 32 + lcol)     = rb0;
            *(uint4*)(sB[nxt] + lrow * 32 + lcol + 8) = rb1;
            __syncthreads();
        }
    }

#pragma unroll
    for (int i = 0; i < 2; ++i)
#pragma unroll
        for (int j = 0; j < 4; ++j)
#pragma unroll
            for (int r = 0; r < 8; ++r) {
                int m = m0 + wm * 32 + i * 16 + r + hi * 8;
                int n = n0 + wn * 64 + j * 16 + nn;
                float bb = bias ? bias[n] : 0.f;
                C[(size_t)m * Nout + n] = acc[i][j][r] + bb;
            }
}

// ---------------------------------------------------------------------------
// Per-(row,head) L2 normalize + reorder [N, D_OUT] f32 -> [B,H,T,Dh] f16
// ---------------------------------------------------------------------------
__launch_bounds__(256)
__global__ void norm_reorder(const float* __restrict__ Y, _Float16* __restrict__ dst, int doNorm) {
    __shared__ float red[256];
    int tid = threadIdx.x;
    int seg = tid >> 6;       // 4 head-rows per block
    int ln  = tid & 63;
    int gid = blockIdx.x * 4 + seg;      // 0 .. N_ROWS*NUM_HEADS-1
    int n = gid >> 4;
    int h = gid & 15;
    float v = Y[(size_t)n * D_OUT + h * HEAD_DIM + ln];
    red[tid] = v * v;
    __syncthreads();
    for (int off = 32; off > 0; off >>= 1) {
        if (ln < off) red[tid] += red[tid + off];
        __syncthreads();
    }
    float nrm = sqrtf(red[seg * 64]);
    float out = doNorm ? (v / fmaxf(nrm, EPS)) : v;
    int b = n / T_SZ, t = n % T_SZ;
    dst[((size_t)(b * NUM_HEADS + h) * T_SZ + t) * HEAD_DIM + ln] = (_Float16)out;
}

// ---------------------------------------------------------------------------
// Angular attention, flash-style: 64 queries / block, 64-key panels staged
// through LDS via async global->LDS loads (ASYNCcnt). Weights are positive
// and row-normalized by a plain sum -> numerator and denominator accumulate
// independently, no online rescaling needed.
// ---------------------------------------------------------------------------
__launch_bounds__(256)
__global__ void angular_attention(const _Float16* __restrict__ Qn,
                                  const _Float16* __restrict__ Kn,
                                  const _Float16* __restrict__ Vh,
                                  _Float16* __restrict__ Ah) {
    __shared__ _Float16 sQ[64 * 64];
    __shared__ _Float16 sK[64 * 64];
    __shared__ _Float16 sV[64 * 64];
    __shared__ _Float16 sW[64 * 64];
    __shared__ float rowsum[64];

    int tid  = threadIdx.x;
    int lane = tid & 31, wave = tid >> 5;
    int wm = wave & 3, wn = wave >> 2;
    int hi = lane >> 4, nn = lane & 15;
    int bh = blockIdx.y;                    // b*H + h
    int q0 = blockIdx.x * 64;
    int b = bh / NUM_HEADS, h = bh % NUM_HEADS;

    const _Float16* Kpanel = Kn + (size_t)bh * T_SZ * HEAD_DIM;
    const _Float16* Vpanel = Vh + (size_t)bh * T_SZ * HEAD_DIM;

    // Q block is fully contiguous in [B,H,T,Dh]: 4096 halfs = 512 uint4
    {
        const uint4* src = (const uint4*)(Qn + ((size_t)bh * T_SZ + q0) * HEAD_DIM);
        ((uint4*)sQ)[tid]       = src[tid];
        ((uint4*)sQ)[tid + 256] = src[tid + 256];
    }
    if (tid < 64) rowsum[tid] = 0.f;

    v8f o0 = {}; v8f o1 = {};
    for (int kb = 0; kb < T_SZ; kb += 64) {
        __syncthreads();   // previous iteration's LDS reads complete
        {
            const _Float16* kg = Kpanel + (size_t)kb * HEAD_DIM;
            const _Float16* vg = Vpanel + (size_t)kb * HEAD_DIM;
            copy_b128_to_lds(kg + tid * 8,         sK + tid * 8);
            copy_b128_to_lds(kg + (tid + 256) * 8, sK + (tid + 256) * 8);
            copy_b128_to_lds(vg + tid * 8,         sV + tid * 8);
            copy_b128_to_lds(vg + (tid + 256) * 8, sV + (tid + 256) * 8);
            if (kb + 64 < T_SZ) {   // L2 prefetch of next panel while async lands
                __builtin_prefetch(kg + 64 * HEAD_DIM + tid * 16, 0, 1);
                __builtin_prefetch(vg + 64 * HEAD_DIM + tid * 16, 0, 1);
            }
            wait_async0();
        }
        __syncthreads();

        // ---- S = Q*K^T, then angular weight transform, write f16 to sW ----
        float rs[8];
#pragma unroll
        for (int r = 0; r < 8; ++r) rs[r] = 0.f;
#pragma unroll
        for (int t = 0; t < 2; ++t) {
            v8f s = {};
#pragma unroll
            for (int kk = 0; kk < HEAD_DIM; kk += 32) {
                v16h a  = load_frag_A(sQ + wm * 16 * 64 + kk, 64);
                v16h bb = load_frag_B_nmaj(sK + (wn * 32 + t * 16) * 64 + kk, 64);
                s = __builtin_amdgcn_wmma_f32_16x16x32_f16(false, a, false, bb, (short)0, s, false, false);
            }
#pragma unroll
            for (int r = 0; r < 8; ++r) {
                float cs  = fminf(fmaxf(s[r], -0.999f), 0.999f);
                float sc  = 1.f - acosf(cs) * INV_PI;
                float w   = fmaxf(sc, EPS);
                float w2  = w * w, w4 = w2 * w2, w8 = w4 * w4;
                float w16 = w8 * w8;
                rs[r] += w16;
                sW[(wm * 16 + r + hi * 8) * 64 + wn * 32 + t * 16 + nn] = (_Float16)w16;
            }
        }
        // row sums: reduce over the 16 lanes of each half, one atomic per row
#pragma unroll
        for (int r = 0; r < 8; ++r) {
            float v = rs[r];
            for (int off = 1; off < 16; off <<= 1) v += __shfl_xor(v, off, 16);
            if (nn == 0) atomicAdd(&rowsum[wm * 16 + r + hi * 8], v);
        }
        __syncthreads();

        // ---- O += W * V ----
#pragma unroll
        for (int kk = 0; kk < 64; kk += 32) {
            v16h a  = load_frag_A(sW + wm * 16 * 64 + kk, 64);
            v16h b0 = load_frag_B_kmaj(sV + kk * 64 + wn * 32 + 0, 64);
            v16h b1 = load_frag_B_kmaj(sV + kk * 64 + wn * 32 + 16, 64);
            o0 = __builtin_amdgcn_wmma_f32_16x16x32_f16(false, a, false, b0, (short)0, o0, false, false);
            o1 = __builtin_amdgcn_wmma_f32_16x16x32_f16(false, a, false, b1, (short)0, o1, false, false);
        }
    }
    __syncthreads();

#pragma unroll
    for (int r = 0; r < 8; ++r) {
        int   m     = wm * 16 + r + hi * 8;
        float denom = rowsum[m] + EPS;
        int   t     = q0 + m;
        size_t rowo = ((size_t)(b * T_SZ) + t) * D_OUT + h * HEAD_DIM + wn * 32;
        Ah[rowo + nn]      = (_Float16)(o0[r] / denom);
        Ah[rowo + 16 + nn] = (_Float16)(o1[r] / denom);
    }
}

// ---------------------------------------------------------------------------
extern "C" void kernel_launch(void* const* d_in, const int* in_sizes, int n_in,
                              void* d_out, int out_size, void* d_ws, size_t ws_size,
                              hipStream_t stream) {
    (void)in_sizes; (void)n_in; (void)out_size; (void)ws_size;
    const float* x  = (const float*)d_in[0];
    const float* Wq = (const float*)d_in[1];
    const float* Wk = (const float*)d_in[2];
    const float* Wv = (const float*)d_in[3];
    const float* Wo = (const float*)d_in[4];
    const float* bo = (const float*)d_in[5];
    float* out = (float*)d_out;

    char* w = (char*)d_ws;
    _Float16* Xh  = (_Float16*)w; w += (size_t)N_ROWS * D_IN * sizeof(_Float16);   // 8 MB
    _Float16* Wqh = (_Float16*)w; w += (size_t)D_OUT * D_IN * sizeof(_Float16);    // 2 MB
    _Float16* Wkh = (_Float16*)w; w += (size_t)D_OUT * D_IN * sizeof(_Float16);
    _Float16* Wvh = (_Float16*)w; w += (size_t)D_OUT * D_IN * sizeof(_Float16);
    _Float16* Woh = (_Float16*)w; w += (size_t)D_OUT * D_OUT * sizeof(_Float16);
    _Float16* Qnh = (_Float16*)w; w += (size_t)N_ROWS * D_OUT * sizeof(_Float16);  // 8 MB each
    _Float16* Knh = (_Float16*)w; w += (size_t)N_ROWS * D_OUT * sizeof(_Float16);
    _Float16* Vhh = (_Float16*)w; w += (size_t)N_ROWS * D_OUT * sizeof(_Float16);
    _Float16* Ah  = (_Float16*)w; w += (size_t)N_ROWS * D_OUT * sizeof(_Float16);
    float*    Yf  = (float*)w;    w += (size_t)N_ROWS * D_OUT * sizeof(float);     // 16 MB

    // f32 -> f16 staging
    cvt_f32_f16<<<1024, 256, 0, stream>>>(x,  Xh,  N_ROWS * D_IN);
    cvt_f32_f16<<<512,  256, 0, stream>>>(Wq, Wqh, D_OUT * D_IN);
    cvt_f32_f16<<<512,  256, 0, stream>>>(Wk, Wkh, D_OUT * D_IN);
    cvt_f32_f16<<<512,  256, 0, stream>>>(Wv, Wvh, D_OUT * D_IN);
    cvt_f32_f16<<<512,  256, 0, stream>>>(Wo, Woh, D_OUT * D_OUT);

    dim3 gemmGrid(D_OUT / 128, N_ROWS / 128);
    int normBlocks = N_ROWS * NUM_HEADS / 4;

    // Q / K / V projections (Yf scratch reused sequentially)
    gemm_f16_wmma<<<gemmGrid, 256, 0, stream>>>(Xh, Wqh, Yf, nullptr, N_ROWS, D_OUT, D_IN);
    norm_reorder<<<normBlocks, 256, 0, stream>>>(Yf, Qnh, 1);
    gemm_f16_wmma<<<gemmGrid, 256, 0, stream>>>(Xh, Wkh, Yf, nullptr, N_ROWS, D_OUT, D_IN);
    norm_reorder<<<normBlocks, 256, 0, stream>>>(Yf, Knh, 1);
    gemm_f16_wmma<<<gemmGrid, 256, 0, stream>>>(Xh, Wvh, Yf, nullptr, N_ROWS, D_OUT, D_IN);
    norm_reorder<<<normBlocks, 256, 0, stream>>>(Yf, Vhh, 0);

    // attention
    dim3 attGrid(T_SZ / 64, B_SZ * NUM_HEADS);
    angular_attention<<<attGrid, 256, 0, stream>>>(Qnh, Knh, Vhh, Ah);

    // output projection + bias -> f32 d_out
    gemm_f16_wmma<<<gemmGrid, 256, 0, stream>>>(Ah, Woh, out, bo, N_ROWS, D_OUT, D_OUT);
}
